// PointClsStem_1520418423422
// MI455X (gfx1250) — matrix operand order repaired and checked by reading, not compiled, not run on hardware
//
#include <hip/hip_runtime.h>
#include <hip/hip_bf16.h>

// ---------------- problem constants (fixed by reference) ----------------
#define BATCH 4
#define NPTS  8192           // points per batch
#define CIN   3
#define DCH   64             // output channels
#define KNN   16             // neighbors (== WMMA M tile!)
#define FDIM  9              // 3*CIN feature channels
#define BN_EPS 1e-5f
#define TOTAL_PTS (BATCH * NPTS)            // 32768  (waves in kernel 2)
#define GEMM_BLOCKS (TOTAL_PTS / 8)         // 4096 blocks of 8 waves

typedef __attribute__((ext_vector_type(16))) _Float16 v16h;
typedef __attribute__((ext_vector_type(8)))  float    v8f;

// =======================================================================
// Kernel 1: streaming KNN. One thread per query point; candidates staged
// through LDS in 256-point tiles; running unsorted top-16 kept in VGPRs
// (replace-worst + unrolled rescan keeps arrays in registers).
// =======================================================================
__global__ void __launch_bounds__(256)
knn_kernel(const float* __restrict__ x, int* __restrict__ idxb) {
  __shared__ float sx[256], sy[256], sz[256];

  const int q = blockIdx.x * 256 + threadIdx.x;   // global query id in [0, B*N)
  const int b = q >> 13;                          // q / NPTS
  const float px = x[(size_t)q * 3 + 0];
  const float py = x[(size_t)q * 3 + 1];
  const float pz = x[(size_t)q * 3 + 2];

  float bd[KNN];
  int   bi[KNN];
#pragma unroll
  for (int s = 0; s < KNN; ++s) { bd[s] = 1e30f; bi[s] = 0; }
  float worst = 1e30f;
  int   worstSlot = 0;

  const float* xb = x + (size_t)b * NPTS * 3;

  for (int t = 0; t < NPTS; t += 256) {
    const int ci = t + threadIdx.x;
    sx[threadIdx.x] = xb[(size_t)ci * 3 + 0];
    sy[threadIdx.x] = xb[(size_t)ci * 3 + 1];
    sz[threadIdx.x] = xb[(size_t)ci * 3 + 2];
    __syncthreads();
    if (t + 256 < NPTS)  // warm next tile into cache (global_prefetch_b8)
      __builtin_prefetch(&xb[(size_t)(t + 256 + threadIdx.x) * 3], 0, 1);
#pragma unroll 4
    for (int j = 0; j < 256; ++j) {
      const float dx = sx[j] - px, dy = sy[j] - py, dz = sz[j] - pz;
      const float d = dx * dx + dy * dy + dz * dz;
      if (d < worst) {
        const int g = t + j;
#pragma unroll
        for (int s = 0; s < KNN; ++s)
          if (s == worstSlot) { bd[s] = d; bi[s] = g; }
        worst = bd[0]; worstSlot = 0;
#pragma unroll
        for (int s = 1; s < KNN; ++s)
          if (bd[s] > worst) { worst = bd[s]; worstSlot = s; }
      }
    }
    __syncthreads();
  }
#pragma unroll
  for (int s = 0; s < KNN; ++s) idxb[(size_t)q * KNN + s] = bi[s];
}

// =======================================================================
// Kernel 2: fused edge-feature + 1x1 conv (WMMA) + per-point max/min over
// k + per-block channel sum/sumsq partials. One wave == one point:
//   A (16x32 f16) = 16 neighbor feature rows, K padded 9->32 with zeros
//   B (32x16 f16) = W column tile (4 tiles cover D=64), bias via C operand
// =======================================================================
__global__ void __launch_bounds__(256)
edge_gemm_stats(const float* __restrict__ x, const float* __restrict__ W,
                const float* __restrict__ bias, const int* __restrict__ idxb,
                float* __restrict__ hmax, float* __restrict__ hmin,
                float* __restrict__ partial) {
  __shared__ float lsum[8][DCH];
  __shared__ float lsq[8][DCH];

  const int tid  = threadIdx.x;
  const int lane = tid & 31;
  const int wv   = tid >> 5;                 // wave in block (0..7)
  const int pt   = blockIdx.x * 8 + wv;      // flat (b*N + n)
  const int b    = pt >> 13;
  const int m    = lane & 15;                // neighbor row this lane feeds
  const int half = lane >> 4;                // 0: K 0-7/16-23, 1: K 8-15/24-31

  // ---- build feature row m = [xc, xn-xc, xn] -------------------------
  const float cx = x[(size_t)pt * 3 + 0];
  const float cy = x[(size_t)pt * 3 + 1];
  const float cz = x[(size_t)pt * 3 + 2];
  const int   nj = idxb[(size_t)pt * KNN + m];
  const float* xn = x + ((size_t)b * NPTS + nj) * 3;
  const float nx0 = xn[0], ny0 = xn[1], nz0 = xn[2];
  float f[FDIM] = { cx, cy, cz, nx0 - cx, ny0 - cy, nz0 - cz, nx0, ny0, nz0 };

  // ---- A fragment: ISA 16-bit A 16x32 layout -------------------------
  v16h a;
#pragma unroll
  for (int s = 0; s < 16; ++s) {
    const int K = (s < 8) ? (half * 8 + s) : (16 + half * 8 + (s - 8));
    a[s] = (K < FDIM) ? (_Float16)f[K] : (_Float16)0.0f;
  }

#pragma unroll
  for (int t = 0; t < 4; ++t) {
    const int col = t * 16 + (lane & 15);    // output channel for this lane

    // B fragment: 32x16, lane-half holds K = half*8 + s (s<8), 16 + half*8 + (s-8) (s>=8), N = lane&15
    v16h bf;
#pragma unroll
    for (int s = 0; s < 16; ++s) {
      const int K = (s < 8) ? (half * 8 + s) : (16 + half * 8 + (s - 8));
      bf[s] = (K < FDIM) ? (_Float16)W[K * DCH + col] : (_Float16)0.0f;
    }

    // bias injected through the accumulator (bias depends on channel only)
    const float bv = bias[col];
    v8f c;
#pragma unroll
    for (int r = 0; r < 8; ++r) c[r] = bv;

    c = __builtin_amdgcn_wmma_f32_16x16x32_f16(
        /*neg_a=*/false, a, /*neg_b=*/false, bf,
        /*c_mod=*/(short)0, c, /*reuse_a=*/false, /*reuse_b=*/false);

    // lane holds rows M = 8*half .. 8*half+7 of channel `col`
    float vmax = c[0], vmin = c[0], vsum = 0.f, vsq = 0.f;
#pragma unroll
    for (int r = 0; r < 8; ++r) {
      const float v = c[r];
      vmax = fmaxf(vmax, v); vmin = fminf(vmin, v);
      vsum += v; vsq += v * v;
    }
    // combine the two row-halves (wave32 xor-16 shuffle)
    vmax = fmaxf(vmax, __shfl_xor(vmax, 16, 32));
    vmin = fminf(vmin, __shfl_xor(vmin, 16, 32));
    vsum +=            __shfl_xor(vsum, 16, 32);
    vsq  +=            __shfl_xor(vsq,  16, 32);

    if (half == 0) {                       // 16 lanes x 4 tiles = 64 channels
      hmax[(size_t)pt * DCH + col] = vmax;
      hmin[(size_t)pt * DCH + col] = vmin;
      lsum[wv][col] = vsum;                // unique slot: no atomics needed
      lsq [wv][col] = vsq;
    }
  }
  __syncthreads();
  if (tid < 2 * DCH) {                     // deterministic block partials
    const int ch = tid & (DCH - 1);
    float acc = 0.f;
#pragma unroll
    for (int w = 0; w < 8; ++w)
      acc += (tid < DCH) ? lsum[w][ch] : lsq[w][ch];
    partial[(size_t)blockIdx.x * (2 * DCH) + tid] = acc;
  }
}

// =======================================================================
// Kernel 3: deterministic tree reduction of block partials -> chstats[128]
// (chstats[0..63]=sum, [64..127]=sumsq). One block per stat column.
// =======================================================================
__global__ void __launch_bounds__(256)
reduce_stats(const float* __restrict__ partial, float* __restrict__ chstats) {
  __shared__ float sm[256];
  const int s = blockIdx.x;                // 0..127
  float acc = 0.f;
  for (int blk = threadIdx.x; blk < GEMM_BLOCKS; blk += 256)
    acc += partial[(size_t)blk * 128 + s];
  sm[threadIdx.x] = acc;
  __syncthreads();
  for (int off = 128; off > 0; off >>= 1) {
    if (threadIdx.x < off) sm[threadIdx.x] += sm[threadIdx.x + off];
    __syncthreads();
  }
  if (threadIdx.x == 0) chstats[s] = sm[0];
}

// =======================================================================
// Kernel 4: BN affine (batch stats) + ReLU + max_k. max over k commutes
// with the monotone per-channel affine: pick hmax if scale>=0 else hmin.
// =======================================================================
__global__ void __launch_bounds__(256)
bn_relu_maxk(const float* __restrict__ hmax, const float* __restrict__ hmin,
             const float* __restrict__ chstats,
             const float* __restrict__ gamma, const float* __restrict__ beta,
             float* __restrict__ out) {
  const int i = blockIdx.x * 256 + threadIdx.x;    // over B*N*D
  if (i >= TOTAL_PTS * DCH) return;
  const int d = i & (DCH - 1);
  const float cnt  = (float)TOTAL_PTS * (float)KNN;
  const float mean = chstats[d] / cnt;
  const float var  = chstats[DCH + d] / cnt - mean * mean;
  const float sc   = gamma[d] * rsqrtf(var + BN_EPS);
  const float sh   = beta[d] - sc * mean;
  const float h    = (sc >= 0.f) ? hmax[i] : hmin[i];
  out[i] = fmaxf(sc * h + sh, 0.f);
}

// =======================================================================
extern "C" void kernel_launch(void* const* d_in, const int* in_sizes, int n_in,
                              void* d_out, int out_size, void* d_ws, size_t ws_size,
                              hipStream_t stream) {
  const float* x     = (const float*)d_in[0];   // (B,N,3)
  const float* W     = (const float*)d_in[1];   // (9,64)
  const float* bias  = (const float*)d_in[2];   // (64)
  const float* gamma = (const float*)d_in[3];   // (64)
  const float* beta  = (const float*)d_in[4];   // (64)
  float* out = (float*)d_out;                   // (B,N,64)

  // workspace carve-up (~20 MB)
  int*   idxb    = (int*)d_ws;                               // B*N*16 ints (2 MB)
  float* hmax    = (float*)(idxb + (size_t)TOTAL_PTS * KNN); // 8 MB
  float* hmin    = hmax + (size_t)TOTAL_PTS * DCH;           // 8 MB
  float* partial = hmin + (size_t)TOTAL_PTS * DCH;           // 4096*128 (2 MB)
  float* chstats = partial + (size_t)GEMM_BLOCKS * 2 * DCH;  // 128 floats

  knn_kernel     <<<TOTAL_PTS / 256, 256, 0, stream>>>(x, idxb);
  edge_gemm_stats<<<GEMM_BLOCKS,     256, 0, stream>>>(x, W, bias, idxb,
                                                       hmax, hmin, partial);
  reduce_stats   <<<2 * DCH,         256, 0, stream>>>(partial, chstats);
  bn_relu_maxk   <<<(TOTAL_PTS * DCH + 255) / 256, 256, 0, stream>>>(
      hmax, hmin, chstats, gamma, beta, out);
}
